// TransformerXLTemporal_91216515433046
// MI455X (gfx1250) — matrix operand order repaired and checked by reading, not compiled, not run on hardware
//
#include <hip/hip_runtime.h>

// ---------------------------------------------------------------------------
// Transformer-XL forward for MI455X (gfx1250): bf16 WMMA + async LDS staging.
// ---------------------------------------------------------------------------

typedef __bf16 bf16_t;
typedef bf16_t v8bf  __attribute__((ext_vector_type(8)));
typedef bf16_t v16bf __attribute__((ext_vector_type(16)));
typedef float  v8f   __attribute__((ext_vector_type(8)));

#define D_MODEL 1024
#define N_HEAD  16
#define HEAD_DIM 64
#define D_INNER 4096
#define N_LAYER 4
#define SEQ     1024
#define BATCH   4
#define MROWS   (BATCH * SEQ)
#define NEGF    (-3.0e38f)

static __device__ __forceinline__ v8f wmma_bf16(v16bf a, v16bf b, v8f c) {
  return __builtin_amdgcn_wmma_f32_16x16x32_bf16(false, a, false, b, (short)0, c,
                                                 false, false);
}

static __device__ __forceinline__ v16bf cat8(v8bf lo, v8bf hi) {
  v16bf r;
#pragma unroll
  for (int i = 0; i < 8; i++) { r[i] = lo[i]; r[i + 8] = hi[i]; }
  return r;
}

// A fragment (16 x 32, bf16, row-major source with given stride):
// lane holds row (lane&15); K-chunks at (lane>>4)*8 and +16.
static __device__ __forceinline__ v16bf load_a_frag(const bf16_t* base, int stride,
                                                    int lane) {
  const bf16_t* p = base + (lane & 15) * stride + ((lane >> 4) * 8);
  return cat8(*(const v8bf*)p, *(const v8bf*)(p + 16));
}

// B fragment (32 x 16): source rows are the 16 "N" columns (e.g. rows of W),
// each K-major. lane holds column (lane&15); K-chunks at (lane>>4)*8 and +16.
static __device__ __forceinline__ v16bf load_b_frag(const bf16_t* base, int stride,
                                                    int lane) {
  const bf16_t* p = base + (lane & 15) * stride + ((lane >> 4) * 8);
  return cat8(*(const v8bf*)p, *(const v8bf*)(p + 16));
}

// Async copy 16 bytes global -> LDS (tracked by ASYNCcnt). LDS byte address is
// the low 32 bits of the generic pointer (flat LDS aperture keeps the
// wave-relative offset in addr[31:0]).
static __device__ __forceinline__ void async_ld_b128(const bf16_t* lds_dst,
                                                     const bf16_t* gsrc) {
  unsigned lds_off = (unsigned)(uintptr_t)lds_dst;
  unsigned long long ga = (unsigned long long)(uintptr_t)gsrc;
  asm volatile("global_load_async_to_lds_b128 %0, %1, off"
               :: "v"(lds_off), "v"(ga)
               : "memory");
}

static __device__ __forceinline__ void wait_async0() {
  asm volatile("s_wait_asynccnt 0x0" ::: "memory");
}

// ---------------------------------------------------------------------------
// Sinusoidal table rr[d][i], d = relative distance q-k in [0, S-1].
// ---------------------------------------------------------------------------
__global__ void rr_kernel(bf16_t* __restrict__ RR) {
  const int d = blockIdx.x, i = threadIdx.x;      // 64 threads
  const int i2 = (i < 32) ? i : i - 32;
  const float inv = __powf(10000.f, -(2.f * i2) / (float)HEAD_DIM);
  const float s = (float)d * inv;
  RR[(size_t)d * HEAD_DIM + i] = (bf16_t)((i < 32) ? __sinf(s) : __cosf(s));
}

// ---------------------------------------------------------------------------
// LayerNorm: fp32 in -> bf16 out, one row (1024) per 256-thread block.
// ---------------------------------------------------------------------------
__global__ __launch_bounds__(256) void ln_kernel(const float* __restrict__ X,
                                                 const float* __restrict__ w,
                                                 const float* __restrict__ b,
                                                 bf16_t* __restrict__ out) {
  __shared__ float red[256];
  const int row = blockIdx.x, t = threadIdx.x;
  const float* xr = X + (size_t)row * D_MODEL;
  float4 v = ((const float4*)xr)[t];
  red[t] = v.x + v.y + v.z + v.w;
  __syncthreads();
#pragma unroll
  for (int o = 128; o > 0; o >>= 1) { if (t < o) red[t] += red[t + o]; __syncthreads(); }
  const float mean = red[0] * (1.f / D_MODEL);
  __syncthreads();
  const float dx = v.x - mean, dy = v.y - mean, dz = v.z - mean, dw = v.w - mean;
  red[t] = dx * dx + dy * dy + dz * dz + dw * dw;
  __syncthreads();
#pragma unroll
  for (int o = 128; o > 0; o >>= 1) { if (t < o) red[t] += red[t + o]; __syncthreads(); }
  const float inv = rsqrtf(red[0] * (1.f / D_MODEL) + 1e-6f);
  const int c = t * 4;
  bf16_t* o4 = out + (size_t)row * D_MODEL + c;
  o4[0] = (bf16_t)(dx * inv * w[c + 0] + b[c + 0]);
  o4[1] = (bf16_t)(dy * inv * w[c + 1] + b[c + 1]);
  o4[2] = (bf16_t)(dz * inv * w[c + 2] + b[c + 2]);
  o4[3] = (bf16_t)(dw * inv * w[c + 3] + b[c + 3]);
}

// ---------------------------------------------------------------------------
// GEMM: C[M,N] = A(bf16 [M,K]) @ W^T (W fp32 [N,K], converted on load) + bias.
// Epilogues: plain bf16 | bf16 + (val+u) bf16 | GELU bf16 | residual fp32.
// 256 threads = 8 waves; block tile 128x128, K-step 32; wave tile 32x64.
// A tile staged global->LDS via GLOBAL_LOAD_ASYNC_TO_LDS_B128 (ASYNCcnt).
// ---------------------------------------------------------------------------
enum { EPI_BF16 = 0, EPI_QU = 1, EPI_GELU = 2, EPI_RES = 3 };

template <int EPI>
__global__ __launch_bounds__(256, 2) void gemm_kernel(
    const bf16_t* __restrict__ A, const float* __restrict__ W,
    const float* __restrict__ bias, const float* __restrict__ u,
    const float* __restrict__ res, bf16_t* __restrict__ outb,
    bf16_t* __restrict__ outb2, float* __restrict__ outf, int M, int N, int K) {
  __shared__ __align__(16) bf16_t As[128 * 32];
  __shared__ __align__(16) bf16_t Bs[128 * 32];
  const int t = threadIdx.x, lane = t & 31, wv = t >> 5;
  const int m0 = blockIdx.y * 128, n0 = blockIdx.x * 128;
  const int mw = (wv & 3) * 32, nw = (wv >> 2) * 64;
  const int lr = t >> 1, lc = (t & 1) * 16;

  v8f acc[2][4] = {};

  for (int kk = 0; kk < K; kk += 32) {
    __syncthreads();  // previous tile's reads complete before overwrite
    {  // A tile: bf16 128x32, async global -> LDS
      const bf16_t* g = A + (size_t)(m0 + lr) * K + kk + lc;
      async_ld_b128(&As[lr * 32 + lc], g);
      async_ld_b128(&As[lr * 32 + lc + 8], g + 8);
    }
    {  // W tile: fp32 -> bf16 128x32 (needs conversion: VGPR path)
      const float* gw = W + (size_t)(n0 + lr) * K + kk + lc;
      if (kk + 32 < K) __builtin_prefetch(gw + 32, 0, 3);
      const float4* g4 = (const float4*)gw;
      float4 f0 = g4[0], f1 = g4[1], f2 = g4[2], f3 = g4[3];
      v8bf b0, b1;
      b0[0]=(bf16_t)f0.x; b0[1]=(bf16_t)f0.y; b0[2]=(bf16_t)f0.z; b0[3]=(bf16_t)f0.w;
      b0[4]=(bf16_t)f1.x; b0[5]=(bf16_t)f1.y; b0[6]=(bf16_t)f1.z; b0[7]=(bf16_t)f1.w;
      b1[0]=(bf16_t)f2.x; b1[1]=(bf16_t)f2.y; b1[2]=(bf16_t)f2.z; b1[3]=(bf16_t)f2.w;
      b1[4]=(bf16_t)f3.x; b1[5]=(bf16_t)f3.y; b1[6]=(bf16_t)f3.z; b1[7]=(bf16_t)f3.w;
      *(v8bf*)&Bs[lr * 32 + lc] = b0;
      *(v8bf*)&Bs[lr * 32 + lc + 8] = b1;
    }
    wait_async0();   // my async LDS writes landed (barrier handles DScnt)
    __syncthreads();
    v16bf af[2], bfr[4];
#pragma unroll
    for (int i = 0; i < 2; i++) af[i] = load_a_frag(&As[(mw + i * 16) * 32], 32, lane);
#pragma unroll
    for (int j = 0; j < 4; j++) bfr[j] = load_b_frag(&Bs[(nw + j * 16) * 32], 32, lane);
#pragma unroll
    for (int i = 0; i < 2; i++)
#pragma unroll
      for (int j = 0; j < 4; j++) acc[i][j] = wmma_bf16(af[i], bfr[j], acc[i][j]);
  }

  const int lg = (lane >> 4) * 8, ln = lane & 15;
#pragma unroll
  for (int i = 0; i < 2; i++) {
#pragma unroll
    for (int j = 0; j < 4; j++) {
      const int n = n0 + nw + j * 16 + ln;
      const float bz = bias[n];
      const float uz = (EPI == EPI_QU) ? u[n] : 0.f;
#pragma unroll
      for (int v = 0; v < 8; v++) {
        const int m = m0 + mw + i * 16 + lg + v;
        const float val = acc[i][j][v] + bz;
        const size_t idx = (size_t)m * N + n;
        if (EPI == EPI_BF16) {
          outb[idx] = (bf16_t)val;
        } else if (EPI == EPI_QU) {
          outb[idx] = (bf16_t)val;
          outb2[idx] = (bf16_t)(val + uz);
        } else if (EPI == EPI_GELU) {
          outb[idx] = (bf16_t)(0.5f * val * (1.f + erff(val * 0.70710678f)));
        } else {
          outf[idx] = res[idx] + val;
        }
      }
    }
  }
}

// ---------------------------------------------------------------------------
// Fused rel-attention (flash style). Block = 4 waves x 32 = 128 threads,
// covering 64 q rows (16 per wave) of one (b,h). K/V streamed in 32-chunks;
// K chunk staged via async global->LDS, V transposed through VGPRs.
// rel[q,k] = (q+u) . rr[q-k] computed as WMMA against a 48-row window of rr,
// gathered through per-wave LDS.
// ---------------------------------------------------------------------------
__global__ __launch_bounds__(128) void attn_kernel(
    const bf16_t* __restrict__ Q, const bf16_t* __restrict__ QU,
    const bf16_t* __restrict__ Kb, const bf16_t* __restrict__ Vb,
    const bf16_t* __restrict__ RR, bf16_t* __restrict__ CTX) {
  __shared__ __align__(16) bf16_t Ks[32 * 64];     // [k][d]
  __shared__ __align__(16) bf16_t VTs[64 * 32];    // [d][k]
  __shared__ __align__(16) float Gs[4][16 * 48];   // per-wave rel window
  __shared__ __align__(16) bf16_t Ps[4][16 * 32];  // per-wave P transpose

  const int t = threadIdx.x, lane = t & 31, wv = t >> 5;
  const int qblk = blockIdx.x;
  const int b = blockIdx.y >> 4, h = blockIdx.y & 15;
  const size_t off = ((size_t)b * SEQ) * D_MODEL + (size_t)h * HEAD_DIM;
  const int q0 = qblk * 64 + wv * 16;
  const int lr = lane & 15, lg = lane >> 4;

  v16bf aq[2], aqu[2];
#pragma unroll
  for (int dc = 0; dc < 2; dc++) {
    const bf16_t* gq = Q + off + (size_t)(q0 + lr) * D_MODEL + dc * 32 + lg * 8;
    aq[dc] = cat8(*(const v8bf*)gq, *(const v8bf*)(gq + 16));
    const bf16_t* gu = QU + off + (size_t)(q0 + lr) * D_MODEL + dc * 32 + lg * 8;
    aqu[dc] = cat8(*(const v8bf*)gu, *(const v8bf*)(gu + 16));
  }

  v8f ctx[4] = {};
  float mrun[8], lrun[8];
#pragma unroll
  for (int v = 0; v < 8; v++) { mrun[v] = NEGF; lrun[v] = 0.f; }

  const int nch = qblk * 2 + 2;
  for (int ch = 0; ch < nch; ch++) {
    const int kc = ch * 32;
    __syncthreads();
    {  // cooperative K chunk (async -> LDS) + transposed V chunk (VGPR path)
      const int kr = t >> 2, c = (t & 3) * 16;
      const bf16_t* gk = Kb + off + (size_t)(kc + kr) * D_MODEL + c;
      async_ld_b128(&Ks[kr * 64 + c], gk);
      async_ld_b128(&Ks[kr * 64 + c + 8], gk + 8);
      const bf16_t* gv = Vb + off + (size_t)(kc + kr) * D_MODEL + c;
      v8bf v0 = *(const v8bf*)gv, v1 = *(const v8bf*)(gv + 8);
#pragma unroll
      for (int j = 0; j < 8; j++) {
        VTs[(c + j) * 32 + kr] = v0[j];
        VTs[(c + 8 + j) * 32 + kr] = v1[j];
      }
    }
    wait_async0();
    __syncthreads();
    if (kc > q0 + 15) continue;  // wave-uniform; no barriers below

    // ---- scores: Q (16x64) @ K^T (64x32) -> two 16x16 C frags
    v8f sc[2] = {};
#pragma unroll
    for (int jt = 0; jt < 2; jt++) {
#pragma unroll
      for (int dc = 0; dc < 2; dc++) {
        const bf16_t* p = &Ks[(jt * 16 + lr) * 64 + dc * 32 + lg * 8];
        v16bf bk = cat8(*(const v8bf*)p, *(const v8bf*)(p + 16));
        sc[jt] = wmma_bf16(aq[dc], bk, sc[jt]);
      }
    }

    // ---- rel window G = (q+u) @ RW^T, window of 48 distances
    const int d0w = q0 - kc - 31;
#pragma unroll
    for (int jt = 0; jt < 3; jt++) {
      int rrow = d0w + jt * 16 + lr;
      rrow = rrow < 0 ? 0 : (rrow > SEQ - 1 ? SEQ - 1 : rrow);
      v8f g = {};
#pragma unroll
      for (int dc = 0; dc < 2; dc++) {
        const bf16_t* p = RR + (size_t)rrow * HEAD_DIM + dc * 32 + lg * 8;
        v16bf br = cat8(*(const v8bf*)p, *(const v8bf*)(p + 16));
        g = wmma_bf16(aqu[dc], br, g);
      }
#pragma unroll
      for (int v = 0; v < 8; v++)
        Gs[wv][(v + lg * 8) * 48 + jt * 16 + lr] = g[v];
    }
    __asm__ volatile("s_wait_dscnt 0" ::: "memory");  // wave-private LDS RAW

    // ---- gather rel, scale, causal mask, online softmax
    float pbuf[2][8], scale[8];
#pragma unroll
    for (int v = 0; v < 8; v++) {
      const int qi = v + lg * 8;
      const int kj0 = lr, kj1 = 16 + lr;
      float s0 = (sc[0][v] + Gs[wv][qi * 48 + (qi - kj0 + 31)]) * 0.125f;
      float s1 = (sc[1][v] + Gs[wv][qi * 48 + (qi - kj1 + 31)]) * 0.125f;
      if (kc + kj0 > q0 + qi) s0 = NEGF;
      if (kc + kj1 > q0 + qi) s1 = NEGF;
      float cmax = fmaxf(s0, s1);
#pragma unroll
      for (int o = 1; o < 16; o <<= 1) cmax = fmaxf(cmax, __shfl_xor(cmax, o, 32));
      const float mn = fmaxf(mrun[v], cmax);
      const float sc_ = __expf(mrun[v] - mn);
      const float p0 = __expf(s0 - mn), p1 = __expf(s1 - mn);
      float rs = p0 + p1;
#pragma unroll
      for (int o = 1; o < 16; o <<= 1) rs += __shfl_xor(rs, o, 32);
      lrun[v] = lrun[v] * sc_ + rs;
      mrun[v] = mn;
      scale[v] = sc_;
      pbuf[0][v] = p0;
      pbuf[1][v] = p1;
    }
#pragma unroll
    for (int nd = 0; nd < 4; nd++)
#pragma unroll
      for (int v = 0; v < 8; v++) ctx[nd][v] *= scale[v];

    // ---- P (C layout) -> LDS -> A-frag, then ctx += P @ V
#pragma unroll
    for (int v = 0; v < 8; v++) {
      const int qi = v + lg * 8;
      Ps[wv][qi * 32 + lr] = (bf16_t)pbuf[0][v];
      Ps[wv][qi * 32 + 16 + lr] = (bf16_t)pbuf[1][v];
    }
    __asm__ volatile("s_wait_dscnt 0" ::: "memory");
    const bf16_t* pp = &Ps[wv][lr * 32 + lg * 8];
    v16bf pf = cat8(*(const v8bf*)pp, *(const v8bf*)(pp + 16));
#pragma unroll
    for (int nd = 0; nd < 4; nd++) {
      const bf16_t* vp = &VTs[(nd * 16 + lr) * 32 + lg * 8];
      v16bf bv = cat8(*(const v8bf*)vp, *(const v8bf*)(vp + 16));
      ctx[nd] = wmma_bf16(pf, bv, ctx[nd]);
    }
  }

  // ---- finalize
#pragma unroll
  for (int nd = 0; nd < 4; nd++) {
#pragma unroll
    for (int v = 0; v < 8; v++) {
      const int qi = v + lg * 8;
      CTX[off + (size_t)(q0 + qi) * D_MODEL + nd * 16 + lr] =
          (bf16_t)(ctx[nd][v] / lrun[v]);
    }
  }
}

// ---------------------------------------------------------------------------
extern "C" void kernel_launch(void* const* d_in, const int* in_sizes, int n_in,
                              void* d_out, int out_size, void* d_ws, size_t ws_size,
                              hipStream_t stream) {
  const float* x   = (const float*)d_in[0];
  const float* Wq  = (const float*)d_in[1];
  const float* bq  = (const float*)d_in[2];
  const float* Wk  = (const float*)d_in[3];
  const float* bk  = (const float*)d_in[4];
  const float* Wv  = (const float*)d_in[5];
  const float* bv  = (const float*)d_in[6];
  const float* Wo  = (const float*)d_in[7];
  const float* bo  = (const float*)d_in[8];
  const float* ub  = (const float*)d_in[9];
  const float* W1  = (const float*)d_in[10];
  const float* b1  = (const float*)d_in[11];
  const float* W2  = (const float*)d_in[12];
  const float* b2  = (const float*)d_in[13];
  const float* l1w = (const float*)d_in[14];
  const float* l1b = (const float*)d_in[15];
  const float* l2w = (const float*)d_in[16];
  const float* l2b = (const float*)d_in[17];
  (void)in_sizes; (void)n_in; (void)out_size; (void)ws_size;

  char* w = (char*)d_ws;
  size_t o = 0;
  auto carve = [&](size_t bytes) -> void* {
    void* p = w + o;
    o += (bytes + 255) & ~(size_t)255;
    return p;
  };
  float*  H    = (float*)carve((size_t)MROWS * D_MODEL * 4);
  bf16_t* XN   = (bf16_t*)carve((size_t)MROWS * D_MODEL * 2);
  bf16_t* Qb   = (bf16_t*)carve((size_t)MROWS * D_MODEL * 2);
  bf16_t* QUb  = (bf16_t*)carve((size_t)MROWS * D_MODEL * 2);
  bf16_t* Kbuf = (bf16_t*)carve((size_t)MROWS * D_MODEL * 2);
  bf16_t* Vbuf = (bf16_t*)carve((size_t)MROWS * D_MODEL * 2);
  bf16_t* CTXb = (bf16_t*)carve((size_t)MROWS * D_MODEL * 2);
  bf16_t* FFb  = (bf16_t*)carve((size_t)MROWS * D_INNER * 2);
  bf16_t* RRb  = (bf16_t*)carve((size_t)SEQ * HEAD_DIM * 2);

  hipMemcpyAsync(H, x, (size_t)MROWS * D_MODEL * 4, hipMemcpyDeviceToDevice, stream);
  rr_kernel<<<SEQ, 64, 0, stream>>>(RRb);

  const dim3 gP(D_MODEL / 128, MROWS / 128);
  const dim3 gF(D_INNER / 128, MROWS / 128);

  for (int l = 0; l < N_LAYER; l++) {
    const float* Wq_l = Wq + (size_t)l * D_MODEL * D_MODEL;
    const float* Wk_l = Wk + (size_t)l * D_MODEL * D_MODEL;
    const float* Wv_l = Wv + (size_t)l * D_MODEL * D_MODEL;
    const float* Wo_l = Wo + (size_t)l * D_MODEL * D_MODEL;
    const float* W1_l = W1 + (size_t)l * D_INNER * D_MODEL;
    const float* W2_l = W2 + (size_t)l * D_MODEL * D_INNER;

    ln_kernel<<<MROWS, 256, 0, stream>>>(H, l1w + l * D_MODEL, l1b + l * D_MODEL, XN);
    gemm_kernel<EPI_QU><<<gP, 256, 0, stream>>>(XN, Wq_l, bq + l * D_MODEL,
        ub + l * N_HEAD * HEAD_DIM, nullptr, Qb, QUb, nullptr, MROWS, D_MODEL, D_MODEL);
    gemm_kernel<EPI_BF16><<<gP, 256, 0, stream>>>(XN, Wk_l, bk + l * D_MODEL,
        nullptr, nullptr, Kbuf, nullptr, nullptr, MROWS, D_MODEL, D_MODEL);
    gemm_kernel<EPI_BF16><<<gP, 256, 0, stream>>>(XN, Wv_l, bv + l * D_MODEL,
        nullptr, nullptr, Vbuf, nullptr, nullptr, MROWS, D_MODEL, D_MODEL);
    attn_kernel<<<dim3(SEQ / 64, BATCH * N_HEAD), 128, 0, stream>>>(
        Qb, QUb, Kbuf, Vbuf, RRb, CTXb);
    gemm_kernel<EPI_RES><<<gP, 256, 0, stream>>>(CTXb, Wo_l, bo + l * D_MODEL,
        nullptr, H, nullptr, nullptr, H, MROWS, D_MODEL, D_MODEL);
    ln_kernel<<<MROWS, 256, 0, stream>>>(H, l2w + l * D_MODEL, l2b + l * D_MODEL, XN);
    gemm_kernel<EPI_GELU><<<gF, 256, 0, stream>>>(XN, W1_l, b1 + l * D_INNER,
        nullptr, nullptr, FFb, nullptr, nullptr, MROWS, D_INNER, D_MODEL);
    gemm_kernel<EPI_RES><<<gP, 256, 0, stream>>>(FFb, W2_l, b2 + l * D_MODEL,
        nullptr, H, nullptr, nullptr, H, MROWS, D_MODEL, D_INNER);
  }

  hipMemcpyAsync(d_out, H, (size_t)MROWS * D_MODEL * 4, hipMemcpyDeviceToDevice,
                 stream);
}
